// CQAttention_77481210020599
// MI455X (gfx1250) — matrix operand order repaired and checked by reading, not compiled, not run on hardware
//
#include <hip/hip_runtime.h>
#include <hip/hip_bf16.h>
#include <math.h>

// ---------------------------------------------------------------------------
// CQAttention for MI455X (gfx1250): bf16 WMMA pipeline, 2x2 register-blocked.
// B=8, Lc=2048, Lq=512, d=512 (fixed by the reference).
// ---------------------------------------------------------------------------

typedef __attribute__((ext_vector_type(16))) __bf16 v16bf;
typedef __attribute__((ext_vector_type(8)))  __bf16 v8bf;
typedef __attribute__((ext_vector_type(8)))  float  v8f;

constexpr int B_   = 8;
constexpr int LC   = 2048;
constexpr int LQ   = 512;
constexpr int D    = 512;
constexpr int OUTW = 4 * D;   // 2048

// ---------------- workspace layout (bytes) ----------------
constexpr size_t OFF_CM16  = 0;                                   // B*LC*D  bf16  (C .* wm)
constexpr size_t OFF_Q16   = OFF_CM16  + (size_t)B_ * LC * D * 2; // B*LQ*D  bf16  (Q row-major)
constexpr size_t OFF_QT16  = OFF_Q16   + (size_t)B_ * LQ * D * 2; // B*D*LQ  bf16  (Q^T)
constexpr size_t OFF_CT16  = OFF_QT16  + (size_t)B_ * D * LQ * 2; // B*D*LC  bf16  (C^T)
constexpr size_t OFF_S     = OFF_CT16  + (size_t)B_ * D * LC * 2; // B*LC*LQ f32
constexpr size_t OFF_SQ16  = OFF_S     + (size_t)B_ * LC * LQ * 4;// B*LC*LQ bf16  (row softmax)
constexpr size_t OFF_SCT16 = OFF_SQ16  + (size_t)B_ * LC * LQ * 2;// B*LQ*LC bf16  (col softmax, transposed)
constexpr size_t OFF_SCTCT = OFF_SCT16 + (size_t)B_ * LQ * LC * 2;// B*D*LQ  bf16  (ScTC^T)
constexpr size_t OFF_RC    = OFF_SCTCT + (size_t)B_ * D * LQ * 2; // B*LC f32
constexpr size_t OFF_RQ    = OFF_RC    + (size_t)B_ * LC * 4;     // B*LQ f32
constexpr size_t OFF_MQ    = OFF_RQ    + (size_t)B_ * LQ * 4;     // B*LC f32 row max
constexpr size_t OFF_SQS   = OFF_MQ    + (size_t)B_ * LC * 4;     // B*LC f32 row sumexp
constexpr size_t OFF_MC    = OFF_SQS   + (size_t)B_ * LC * 4;     // B*LQ f32 col max
constexpr size_t OFF_SCS   = OFF_MC    + (size_t)B_ * LQ * 4;     // B*LQ f32 col sumexp

// ---------------- fragment loads (per ISA 7.12.2 16-bit layouts) ------------
union Frag16 { v16bf v; v8bf h[2]; };

// A fragment 16x32: lane-half h holds K in [8h,8h+8) U [16+8h,16+8h+8)
__device__ inline v16bf load_frag_a(const __bf16* base, int ld, int row0, int k0, int lane) {
    int r = lane & 15, h = lane >> 4;
    const __bf16* p = base + (size_t)(row0 + r) * ld + k0 + h * 8;
    Frag16 f;
    f.h[0] = *(const v8bf*)(p);
    f.h[1] = *(const v8bf*)(p + 16);
    return f.v;
}

// B fragment 32x16 from N-major storage (row n holds K contiguous):
// lane-half h holds K in [8h,8h+8) U [16+8h,16+8h+8)
__device__ inline v16bf load_frag_b(const __bf16* baseT, int ld, int n0, int k0, int lane) {
    int n = lane & 15, h = lane >> 4;
    const __bf16* p = baseT + (size_t)(n0 + n) * ld + k0 + h * 8;
    Frag16 f;
    f.h[0] = *(const v8bf*)(p);
    f.h[1] = *(const v8bf*)(p + 16);
    return f.v;
}

__device__ inline v8f wmma_bf16(v16bf a, v16bf b, v8f c) {
    return __builtin_amdgcn_wmma_f32_16x16x32_bf16(false, a, false, b,
                                                   (short)0, c, false, false);
}

// ---------------- block reductions ----------------
__device__ inline float block_reduce_sum(float v, float* sdata) {
    int t = threadIdx.x;
    sdata[t] = v; __syncthreads();
    for (int s = 128; s > 0; s >>= 1) {
        if (t < s) sdata[t] += sdata[t + s];
        __syncthreads();
    }
    float r = sdata[0]; __syncthreads();
    return r;
}
__device__ inline float block_reduce_max(float v, float* sdata) {
    int t = threadIdx.x;
    sdata[t] = v; __syncthreads();
    for (int s = 128; s > 0; s >>= 1) {
        if (t < s) sdata[t] = fmaxf(sdata[t], sdata[t + s]);
        __syncthreads();
    }
    float r = sdata[0]; __syncthreads();
    return r;
}

// ---------------- pass 1: prep C rows ----------------
// rC = C . wc ; Cm16 = bf16(C .* wm) ; CT16 = bf16(C)^T
__global__ __launch_bounds__(256) void prep_c_kernel(const float* __restrict__ C,
                                                     const float* __restrict__ W,
                                                     float* __restrict__ rC,
                                                     __bf16* __restrict__ Cm16,
                                                     __bf16* __restrict__ CT16) {
    __shared__ float sdata[256];
    int row = blockIdx.x;            // b*LC + i
    int b = row / LC, i = row % LC;
    int t = threadIdx.x;
    const float* Crow = C + (size_t)row * D;
    const float* wc = W;
    const float* wm = W + 2 * D;
    float c0 = Crow[t], c1 = Crow[t + 256];
    float part = c0 * wc[t] + c1 * wc[t + 256];
    Cm16[(size_t)row * D + t]       = (__bf16)(c0 * wm[t]);
    Cm16[(size_t)row * D + t + 256] = (__bf16)(c1 * wm[t + 256]);
    __bf16* CTb = CT16 + (size_t)b * D * LC;
    CTb[(size_t)t * LC + i]         = (__bf16)c0;
    CTb[(size_t)(t + 256) * LC + i] = (__bf16)c1;
    float dot = block_reduce_sum(part, sdata);
    if (t == 0) rC[row] = dot;
}

// ---------------- pass 2: prep Q rows ----------------
__global__ __launch_bounds__(256) void prep_q_kernel(const float* __restrict__ Q,
                                                     const float* __restrict__ W,
                                                     float* __restrict__ rQ,
                                                     __bf16* __restrict__ Q16,
                                                     __bf16* __restrict__ QT16) {
    __shared__ float sdata[256];
    int row = blockIdx.x;            // b*LQ + j
    int b = row / LQ, j = row % LQ;
    int t = threadIdx.x;
    const float* Qrow = Q + (size_t)row * D;
    const float* wq = W + D;
    float q0 = Qrow[t], q1 = Qrow[t + 256];
    float part = q0 * wq[t] + q1 * wq[t + 256];
    Q16[(size_t)row * D + t]       = (__bf16)q0;
    Q16[(size_t)row * D + t + 256] = (__bf16)q1;
    __bf16* QTb = QT16 + (size_t)b * D * LQ;
    QTb[(size_t)t * LQ + j]         = (__bf16)q0;
    QTb[(size_t)(t + 256) * LQ + j] = (__bf16)q1;
    float dot = block_reduce_sum(part, sdata);
    if (t == 0) rQ[row] = dot;
}

// ---------------- pass 3: S = (C.*wm) @ Q^T + rC + rQ + bias ----------------
// 32x32 output tile per wave: 2 A-frags x 2 B-frags -> 4 WMMAs per K-step.
__global__ __launch_bounds__(256) void gemm_s_kernel(const __bf16* __restrict__ Cm16,
                                                     const __bf16* __restrict__ Q16,
                                                     const float* __restrict__ rC,
                                                     const float* __restrict__ rQ,
                                                     const float* __restrict__ bias,
                                                     float* __restrict__ S) {
    constexpr int TI = LC / 32;      // 64
    constexpr int TJ = LQ / 32;      // 16
    int wave = threadIdx.x >> 5, lane = threadIdx.x & 31;
    int tile = blockIdx.x * 8 + wave;
    int b   = tile / (TI * TJ);
    int rem = tile % (TI * TJ);
    int ti = rem / TJ, tj = rem % TJ;
    int row0 = ti * 32, col0 = tj * 32;

    const __bf16* Ab = Cm16 + (size_t)b * LC * D;
    const __bf16* Bb = Q16  + (size_t)b * LQ * D;
    v8f acc00 = {}, acc01 = {}, acc10 = {}, acc11 = {};
    for (int k = 0; k < D; k += 32) {
        v16bf a0 = load_frag_a(Ab, D, row0,      k, lane);
        v16bf a1 = load_frag_a(Ab, D, row0 + 16, k, lane);
        v16bf b0 = load_frag_b(Bb, D, col0,      k, lane);
        v16bf b1 = load_frag_b(Bb, D, col0 + 16, k, lane);
        acc00 = wmma_bf16(a0, b0, acc00);
        acc01 = wmma_bf16(a0, b1, acc01);
        acc10 = wmma_bf16(a1, b0, acc10);
        acc11 = wmma_bf16(a1, b1, acc11);
    }
    float bv = bias[0];
    int h = lane >> 4;
    float* Sb = S + ((size_t)b * LC + row0) * LQ;
    v8f accs[2][2] = {{acc00, acc01}, {acc10, acc11}};
    #pragma unroll
    for (int mi = 0; mi < 2; mi++) {
        #pragma unroll
        for (int nj = 0; nj < 2; nj++) {
            int col = col0 + nj * 16 + (lane & 15);
            float rqv = rQ[b * LQ + col] + bv;
            #pragma unroll
            for (int v = 0; v < 8; v++) {
                int row = mi * 16 + v + 8 * h;
                Sb[(size_t)row * LQ + col] =
                    accs[mi][nj][v] + rC[b * LC + row0 + row] + rqv;
            }
        }
    }
}

// ---------------- pass 4a: row softmax stats (axis j) ----------------
__global__ __launch_bounds__(256) void row_stats_kernel(const float* __restrict__ S,
                                                        float* __restrict__ mq,
                                                        float* __restrict__ sq) {
    __shared__ float sdata[256];
    int row = blockIdx.x;            // b*LC + i
    int t = threadIdx.x;
    const float* Srow = S + (size_t)row * LQ;
    float v0 = Srow[t], v1 = Srow[t + 256];
    float m = block_reduce_max(fmaxf(v0, v1), sdata);
    float e = __expf(v0 - m) + __expf(v1 - m);
    float s = block_reduce_sum(e, sdata);
    if (t == 0) { mq[row] = m; sq[row] = s; }
}

// ---------------- pass 4b: column softmax stats (axis i) ----------------
__global__ __launch_bounds__(256) void col_stats_kernel(const float* __restrict__ S,
                                                        float* __restrict__ mc,
                                                        float* __restrict__ sc) {
    __shared__ float sdata[256];
    int col = blockIdx.x;            // b*LQ + j
    int b = col / LQ, j = col % LQ;
    int t = threadIdx.x;
    float vals[LC / 256];
    float m = -INFINITY;
    #pragma unroll
    for (int r = 0; r < LC / 256; r++) {
        vals[r] = S[((size_t)b * LC + t + r * 256) * LQ + j];
        m = fmaxf(m, vals[r]);
    }
    m = block_reduce_max(m, sdata);
    float e = 0.f;
    #pragma unroll
    for (int r = 0; r < LC / 256; r++) e += __expf(vals[r] - m);
    float s = block_reduce_sum(e, sdata);
    if (t == 0) { mc[col] = m; sc[col] = s; }
}

// ---------------- pass 5: normalize -> Sq16 (row-major), ScT16 (transposed) --
__global__ __launch_bounds__(256) void normalize_kernel(const float* __restrict__ S,
                                                        const float* __restrict__ mq,
                                                        const float* __restrict__ sq,
                                                        const float* __restrict__ mc,
                                                        const float* __restrict__ sc,
                                                        __bf16* __restrict__ Sq16,
                                                        __bf16* __restrict__ ScT16) {
    int row = blockIdx.x;            // b*LC + i
    int b = row / LC, i = row % LC;
    int t = threadIdx.x;
    float mrow = mq[row], inv_row = 1.f / sq[row];
    #pragma unroll
    for (int e = 0; e < 2; e++) {
        int j = t + e * 256;
        float s = S[(size_t)row * LQ + j];
        Sq16[(size_t)row * LQ + j] = (__bf16)(__expf(s - mrow) * inv_row);
        int col = b * LQ + j;
        ScT16[(size_t)col * LC + i] = (__bf16)(__expf(s - mc[col]) / sc[col]);
    }
}

// ---------------- pass 6: ScTC^T = C^T @ Sc  (M=d', N=j, K=i), 32x32 tiles ---
__global__ __launch_bounds__(256) void gemm_sctc_kernel(const __bf16* __restrict__ CT16,
                                                        const __bf16* __restrict__ ScT16,
                                                        __bf16* __restrict__ ScTCT) {
    constexpr int TM = D / 32;       // 16
    constexpr int TN = LQ / 32;      // 16
    int wave = threadIdx.x >> 5, lane = threadIdx.x & 31;
    int tile = blockIdx.x * 8 + wave;
    int b   = tile / (TM * TN);
    int rem = tile % (TM * TN);
    int tm = rem / TN, tn = rem % TN;
    int row0 = tm * 32, col0 = tn * 32;

    const __bf16* Ab = CT16  + (size_t)b * D * LC;   // d x Lc, ld=LC
    const __bf16* Bb = ScT16 + (size_t)b * LQ * LC;  // Lq x Lc (N-major), ld=LC
    v8f acc00 = {}, acc01 = {}, acc10 = {}, acc11 = {};
    for (int k = 0; k < LC; k += 32) {
        v16bf a0 = load_frag_a(Ab, LC, row0,      k, lane);
        v16bf a1 = load_frag_a(Ab, LC, row0 + 16, k, lane);
        v16bf b0 = load_frag_b(Bb, LC, col0,      k, lane);
        v16bf b1 = load_frag_b(Bb, LC, col0 + 16, k, lane);
        acc00 = wmma_bf16(a0, b0, acc00);
        acc01 = wmma_bf16(a0, b1, acc01);
        acc10 = wmma_bf16(a1, b0, acc10);
        acc11 = wmma_bf16(a1, b1, acc11);
    }
    __bf16* Ob = ScTCT + (size_t)b * D * LQ;         // d x Lq, ld=LQ
    int h = lane >> 4;
    v8f accs[2][2] = {{acc00, acc01}, {acc10, acc11}};
    #pragma unroll
    for (int mi = 0; mi < 2; mi++) {
        #pragma unroll
        for (int nj = 0; nj < 2; nj++) {
            int col = col0 + nj * 16 + (lane & 15);
            #pragma unroll
            for (int v = 0; v < 8; v++) {
                int row = row0 + mi * 16 + v + 8 * h;
                Ob[(size_t)row * LQ + col] = (__bf16)accs[mi][nj][v];
            }
        }
    }
}

// ---------------- pass 7: A = Sq@Q, Bm = Sq@ScTC, fused epilogue -------------
// 32 rows x 16 cols x 2 output matrices per wave: 2 A-frags shared across the
// Q and ScTC B-frags -> 4 WMMAs per K-step on 8 loads.
__global__ __launch_bounds__(256) void gemm_out_kernel(const __bf16* __restrict__ Sq16,
                                                       const __bf16* __restrict__ QT16,
                                                       const __bf16* __restrict__ ScTCT,
                                                       const float* __restrict__ C,
                                                       float* __restrict__ out) {
    constexpr int TI = LC / 32;      // 64
    constexpr int TD = D / 16;       // 32
    int wave = threadIdx.x >> 5, lane = threadIdx.x & 31;
    int tile = blockIdx.x * 8 + wave;
    int b   = tile / (TI * TD);
    int rem = tile % (TI * TD);
    int ti = rem / TD, td = rem % TD;
    int row0 = ti * 32;

    const __bf16* Ab  = Sq16  + (size_t)b * LC * LQ; // Lc x Lq, ld=LQ
    const __bf16* Bq  = QT16  + (size_t)b * D * LQ;  // d x Lq (N-major), ld=LQ
    const __bf16* Bsc = ScTCT + (size_t)b * D * LQ;  // d x Lq (N-major), ld=LQ
    v8f accA0 = {}, accA1 = {}, accB0 = {}, accB1 = {};
    for (int k = 0; k < LQ; k += 32) {
        v16bf a0 = load_frag_a(Ab, LQ, row0,      k, lane);
        v16bf a1 = load_frag_a(Ab, LQ, row0 + 16, k, lane);
        v16bf b0 = load_frag_b(Bq,  LQ, td * 16, k, lane);
        v16bf b1 = load_frag_b(Bsc, LQ, td * 16, k, lane);
        accA0 = wmma_bf16(a0, b0, accA0);
        accA1 = wmma_bf16(a1, b0, accA1);
        accB0 = wmma_bf16(a0, b1, accB0);
        accB1 = wmma_bf16(a1, b1, accB1);
    }
    int dcol = td * 16 + (lane & 15);
    int h = lane >> 4;
    v8f accAs[2] = {accA0, accA1};
    v8f accBs[2] = {accB0, accB1};
    #pragma unroll
    for (int mi = 0; mi < 2; mi++) {
        #pragma unroll
        for (int v = 0; v < 8; v++) {
            int i = row0 + mi * 16 + v + 8 * h;
            float c  = C[((size_t)b * LC + i) * D + dcol];
            float a  = accAs[mi][v];
            float bm = accBs[mi][v];
            float* o = out + ((size_t)b * LC + i) * OUTW;
            o[dcol]         = c;
            o[D + dcol]     = a;
            o[2 * D + dcol] = c * a;
            o[3 * D + dcol] = c * bm;
        }
    }
}

// ---------------------------------------------------------------------------
extern "C" void kernel_launch(void* const* d_in, const int* in_sizes, int n_in,
                              void* d_out, int out_size, void* d_ws, size_t ws_size,
                              hipStream_t stream) {
    const float* C    = (const float*)d_in[0];
    const float* Q    = (const float*)d_in[1];
    const float* Wo_w = (const float*)d_in[2];
    const float* Wo_b = (const float*)d_in[3];
    float* out = (float*)d_out;

    char* ws = (char*)d_ws;
    __bf16* Cm16  = (__bf16*)(ws + OFF_CM16);
    __bf16* Q16   = (__bf16*)(ws + OFF_Q16);
    __bf16* QT16  = (__bf16*)(ws + OFF_QT16);
    __bf16* CT16  = (__bf16*)(ws + OFF_CT16);
    float*  S     = (float*) (ws + OFF_S);
    __bf16* Sq16  = (__bf16*)(ws + OFF_SQ16);
    __bf16* ScT16 = (__bf16*)(ws + OFF_SCT16);
    __bf16* ScTCT = (__bf16*)(ws + OFF_SCTCT);
    float*  rC    = (float*) (ws + OFF_RC);
    float*  rQ    = (float*) (ws + OFF_RQ);
    float*  mq    = (float*) (ws + OFF_MQ);
    float*  sq    = (float*) (ws + OFF_SQS);
    float*  mc    = (float*) (ws + OFF_MC);
    float*  sc    = (float*) (ws + OFF_SCS);

    prep_c_kernel<<<B_ * LC, 256, 0, stream>>>(C, Wo_w, rC, Cm16, CT16);
    prep_q_kernel<<<B_ * LQ, 256, 0, stream>>>(Q, Wo_w, rQ, Q16, QT16);

    int tilesS = B_ * (LC / 32) * (LQ / 32);            // 8192
    gemm_s_kernel<<<tilesS / 8, 256, 0, stream>>>(Cm16, Q16, rC, rQ, Wo_b, S);

    row_stats_kernel<<<B_ * LC, 256, 0, stream>>>(S, mq, sq);
    col_stats_kernel<<<B_ * LQ, 256, 0, stream>>>(S, mc, sc);
    normalize_kernel<<<B_ * LC, 256, 0, stream>>>(S, mq, sq, mc, sc, Sq16, ScT16);

    int tilesT = B_ * (D / 32) * (LQ / 32);             // 2048
    gemm_sctc_kernel<<<tilesT / 8, 256, 0, stream>>>(CT16, ScT16, ScTCT);

    int tilesO = B_ * (LC / 32) * (D / 16);             // 16384
    gemm_out_kernel<<<tilesO / 8, 256, 0, stream>>>(Sq16, QT16, ScTCT, C, out);
}
